// GATModel_36507222016234
// MI455X (gfx1250) — matrix operand order repaired and checked by reading, not compiled, not run on hardware
//
#include <hip/hip_runtime.h>

// ---------------------------------------------------------------- types
typedef __attribute__((ext_vector_type(16))) __bf16       v16bf;
typedef __attribute__((ext_vector_type(8)))  float        v8f;
typedef __attribute__((ext_vector_type(4)))  unsigned int v4u;

struct Frag32B { v4u lo, hi; };   // 32 bytes == one v16bf fragment

#define IN_F   128
#define HID_F  64
#define HEADS  4
#define H1_F   256   // HID*HEADS
#define OUT_F  64

// ---------------------------------------------------------------- utility kernels
__global__ void k_f32_to_bf16(const float* __restrict__ in, __bf16* __restrict__ out, int n) {
    int i = blockIdx.x * blockDim.x + threadIdx.x;
    if (i < n) out[i] = (__bf16)in[i];
}

__global__ void k_fill_f32(float* __restrict__ p, float v, int n) {
    int i = blockIdx.x * blockDim.x + threadIdx.x;
    if (i < n) p[i] = v;
}

__global__ void k_build_edges(const int* __restrict__ ei, int* __restrict__ srcF,
                              int* __restrict__ dstF, int E, int N) {
    int i = blockIdx.x * blockDim.x + threadIdx.x;
    int T = E + N;
    if (i >= T) return;
    if (i < E) { srcF[i] = ei[i]; dstF[i] = ei[E + i]; }
    else       { srcF[i] = i - E; dstF[i] = i - E; }
}

// Pack row-major f32 W[K,Nc] into bf16 B-fragment layout:
// Wp[(((kb*tilesN + tn)*32 + lane)*16 + e)] = W[(kb*32 + (lane>>4)*16 + e)*Nc + tn*16 + (lane&15)]
__global__ void k_pack_w_bf16(const float* __restrict__ W, __bf16* __restrict__ Wp,
                              int K, int Nc) {
    int i = blockIdx.x * blockDim.x + threadIdx.x;
    if (i >= K * Nc) return;
    const int e      = i & 15;
    const int lane   = (i >> 4) & 31;
    const int t      = i >> 9;               // kb*tilesN + tn
    const int tilesN = Nc >> 4;
    const int kb = t / tilesN, tn = t % tilesN;
    const int g = lane >> 4, lm = lane & 15;
    const int k = kb * 32 + g * 16 + e;
    const int n = tn * 16 + lm;
    Wp[i] = (__bf16)W[(size_t)k * Nc + n];
}

// ---------------------------------------------------------------- bf16 WMMA GEMM
// C[M,Nc] = A[M,K](bf16 row-major) @ Wp(packed) [+ bias], f32 accumulate.
// One wave per 16x64 C tile: per 32-K step, 1 A fragment (2xb128) feeds 4 WMMAs
// against 4 B fragments (8xb128). Requires Nc % 64 == 0.
__global__ void k_gemm_wmma_bf16(const __bf16* __restrict__ A,
                                 const __bf16* __restrict__ Bp,
                                 const float*  __restrict__ bias,   // may be null
                                 float* __restrict__ C,
                                 int M, int K, int Nc) {
    const int wave    = blockIdx.x * (blockDim.x >> 5) + (threadIdx.x >> 5);
    const int lane    = threadIdx.x & 31;
    const int tilesN  = Nc >> 4;   // 16-col tiles
    const int tilesN4 = Nc >> 6;   // 64-col super-tiles
    const int tilesM  = (M + 15) >> 4;
    if (wave >= tilesM * tilesN4) return;              // wave-uniform exit
    const int tileM  = wave / tilesN4;
    const int tileN4 = wave % tilesN4;

    const int lm = lane & 15;
    const int g  = lane >> 4;
    int m = tileM * 16 + lm;
    if (m >= M) m = M - 1;                 // clamp: loads always in-range
    const int nbase = tileN4 * 64;

    const __bf16* arow = A + (size_t)m * K;
    const int nkb = K >> 5;
    // packed B: 32B per (k-block, 16-col tile, lane); k-block stride = tilesN*512 elems
    const __bf16* bptr  = Bp + (((size_t)tileN4 * 4) * 32 + lane) * 16;
    const size_t  bstep = (size_t)tilesN * 32 * 16;

    v8f acc0 = {}, acc1 = {}, acc2 = {}, acc3 = {};
    for (int kb = 0; kb < nkb; ++kb) {
        Frag32B fa, fb0, fb1, fb2, fb3;
        const int k0 = kb * 32;
        fa.lo  = *(const v4u*)(arow + k0 + g * 16);
        fa.hi  = *(const v4u*)(arow + k0 + g * 16 + 8);
        fb0.lo = *(const v4u*)(bptr +   0);
        fb0.hi = *(const v4u*)(bptr +   8);
        fb1.lo = *(const v4u*)(bptr + 512);
        fb1.hi = *(const v4u*)(bptr + 520);
        fb2.lo = *(const v4u*)(bptr + 1024);
        fb2.hi = *(const v4u*)(bptr + 1032);
        fb3.lo = *(const v4u*)(bptr + 1536);
        fb3.hi = *(const v4u*)(bptr + 1544);
        bptr += bstep;
        const v16bf a  = __builtin_bit_cast(v16bf, fa);
        const v16bf b0 = __builtin_bit_cast(v16bf, fb0);
        const v16bf b1 = __builtin_bit_cast(v16bf, fb1);
        const v16bf b2 = __builtin_bit_cast(v16bf, fb2);
        const v16bf b3 = __builtin_bit_cast(v16bf, fb3);
        acc0 = __builtin_amdgcn_wmma_f32_16x16x32_bf16(false, a, false, b0, (short)0, acc0, false, false);
        acc1 = __builtin_amdgcn_wmma_f32_16x16x32_bf16(false, a, false, b1, (short)0, acc1, false, false);
        acc2 = __builtin_amdgcn_wmma_f32_16x16x32_bf16(false, a, false, b2, (short)0, acc2, false, false);
        acc3 = __builtin_amdgcn_wmma_f32_16x16x32_bf16(false, a, false, b3, (short)0, acc3, false, false);
    }
    #pragma unroll
    for (int j = 0; j < 4; ++j) {
        const v8f acc = (j == 0) ? acc0 : (j == 1) ? acc1 : (j == 2) ? acc2 : acc3;
        const int n = nbase + j * 16 + lm;
        const float bb = bias ? bias[n] : 0.0f;
        #pragma unroll
        for (int r = 0; r < 8; ++r) {
            const int row = tileM * 16 + r + 8 * g;
            if (row < M) C[(size_t)row * Nc + n] = acc[r] + bb;
        }
    }
}

// ---------------------------------------------------------------- attention scores
__global__ void k_scores(const float* __restrict__ xw,
                         const float* __restrict__ a_s, const float* __restrict__ a_d,
                         float* __restrict__ s_src, float* __restrict__ s_dst,
                         int N, int H, int C) {
    int i = blockIdx.x * blockDim.x + threadIdx.x;
    if (i >= N * H) return;
    const int n = i / H, h = i % H;
    const float* row = xw + (size_t)n * H * C + (size_t)h * C;
    float ss = 0.f, sd = 0.f;
    for (int c = 0; c < C; ++c) {
        const float v = row[c];
        ss += v * a_s[h * C + c];
        sd += v * a_d[h * C + c];
    }
    s_src[i] = ss; s_dst[i] = sd;
}

__device__ inline void atomicMaxF(float* addr, float val) {
    if (val >= 0.0f) atomicMax((int*)addr, __float_as_int(val));
    else             atomicMin((unsigned int*)addr, (unsigned int)__float_as_int(val));
}

__global__ void k_edge_alpha(const int* __restrict__ srcF, const int* __restrict__ dstF,
                             const float* __restrict__ s_src, const float* __restrict__ s_dst,
                             float* __restrict__ alpha, float* __restrict__ m,
                             int Etot, int H) {
    int i = blockIdx.x * blockDim.x + threadIdx.x;
    if (i >= Etot * H) return;
    const int e = i / H, h = i % H;
    const int s = srcF[e], d = dstF[e];
    float a = s_src[s * H + h] + s_dst[d * H + h];
    a = (a > 0.0f) ? a : 0.2f * a;
    alpha[i] = a;
    atomicMaxF(&m[d * H + h], a);
}

__global__ void k_edge_exp(const int* __restrict__ dstF,
                           float* __restrict__ alpha, const float* __restrict__ m,
                           float* __restrict__ denom, int Etot, int H) {
    int i = blockIdx.x * blockDim.x + threadIdx.x;
    if (i >= Etot * H) return;
    const int e = i / H, h = i % H;
    const int d = dstF[e];
    const float w = __expf(alpha[i] - m[d * H + h]);
    alpha[i] = w;
    atomicAdd(&denom[d * H + h], w);
}

// one wave per edge: acc[dst,:] += (e/denom[dst]) * xw[src,:]
__global__ void k_edge_scatter(const int* __restrict__ srcF, const int* __restrict__ dstF,
                               const float* __restrict__ alpha, const float* __restrict__ denom,
                               const float* __restrict__ xw, float* __restrict__ acc,
                               int Etot, int H, int C) {
    const int wid  = (blockIdx.x * blockDim.x + threadIdx.x) >> 5;
    const int lane = threadIdx.x & 31;
    if (wid >= Etot) return;
    const int s = srcF[wid], d = dstF[wid];
    const int F = H * C;
    for (int f = lane; f < F; f += 32) {
        const int h = f / C;
        const float w = alpha[(size_t)wid * H + h] / denom[(size_t)d * H + h];
        atomicAdd(&acc[(size_t)d * F + f], w * xw[(size_t)s * F + f]);
    }
}

__global__ void k_bias_relu(const float* __restrict__ acc, const float* __restrict__ b,
                            float* __restrict__ outF32, __bf16* __restrict__ outBf,
                            int total, int F) {
    int i = blockIdx.x * blockDim.x + threadIdx.x;
    if (i >= total) return;
    float v = acc[i] + b[i % F];
    v = v > 0.0f ? v : 0.0f;
    if (outF32) outF32[i] = v;
    outBf[i] = (__bf16)v;
}

// one wave per row: log_softmax over F columns
__global__ void k_log_softmax(const float* __restrict__ logits, float* __restrict__ out,
                              int N, int F) {
    const int wid  = (blockIdx.x * blockDim.x + threadIdx.x) >> 5;
    const int lane = threadIdx.x & 31;
    if (wid >= N) return;
    const float* row = logits + (size_t)wid * F;
    float mx = -3.0e38f;
    for (int f = lane; f < F; f += 32) mx = fmaxf(mx, row[f]);
    #pragma unroll
    for (int o = 16; o > 0; o >>= 1) mx = fmaxf(mx, __shfl_xor(mx, o, 32));
    float se = 0.f;
    for (int f = lane; f < F; f += 32) se += __expf(row[f] - mx);
    #pragma unroll
    for (int o = 16; o > 0; o >>= 1) se += __shfl_xor(se, o, 32);
    const float lse = mx + __logf(se);
    for (int f = lane; f < F; f += 32) out[(size_t)wid * F + f] = row[f] - lse;
}

// ---------------------------------------------------------------- host orchestration
static inline unsigned gblk(long long t, int b) { return (unsigned)((t + b - 1) / b); }

extern "C" void kernel_launch(void* const* d_in, const int* in_sizes, int n_in,
                              void* d_out, int out_size, void* d_ws, size_t ws_size,
                              hipStream_t stream) {
    const float* x   = (const float*)d_in[0];
    const int*   ei  = (const int*)  d_in[1];
    const float* W1  = (const float*)d_in[2];
    const float* as1 = (const float*)d_in[3];
    const float* ad1 = (const float*)d_in[4];
    const float* b1  = (const float*)d_in[5];
    const float* W2  = (const float*)d_in[6];
    const float* as2 = (const float*)d_in[7];
    const float* ad2 = (const float*)d_in[8];
    const float* b2  = (const float*)d_in[9];
    const float* W3  = (const float*)d_in[10];
    const float* as3 = (const float*)d_in[11];
    const float* ad3 = (const float*)d_in[12];
    const float* b3  = (const float*)d_in[13];
    const float* Wc  = (const float*)d_in[14];
    const float* bc  = (const float*)d_in[15];

    const int N    = in_sizes[0] / IN_F;
    const int E    = in_sizes[1] / 2;
    const int Etot = E + N;

    // workspace carve-up (256B aligned)
    char*  base = (char*)d_ws;
    size_t off  = 0;
    auto carve = [&](size_t bytes) -> void* {
        void* p = base + off;
        off = (off + bytes + 255) & ~(size_t)255;
        return p;
    };
    int*    srcF  = (int*)   carve((size_t)Etot * 4);
    int*    dstF  = (int*)   carve((size_t)Etot * 4);
    __bf16* W1b   = (__bf16*)carve((size_t)IN_F * H1_F  * 2);
    __bf16* W2b   = (__bf16*)carve((size_t)H1_F * H1_F  * 2);
    __bf16* W3b   = (__bf16*)carve((size_t)H1_F * HID_F * 2);
    __bf16* Wcb   = (__bf16*)carve((size_t)HID_F * OUT_F * 2);
    __bf16* Abf   = (__bf16*)carve((size_t)N * H1_F * 2);
    float*  xw    = (float*) carve((size_t)N * H1_F * 4);
    float*  sS    = (float*) carve((size_t)N * HEADS * 4);
    float*  sD    = (float*) carve((size_t)N * HEADS * 4);
    float*  mMax  = (float*) carve((size_t)N * HEADS * 4);
    float*  den   = (float*) carve((size_t)N * HEADS * 4);
    float*  alpha = (float*) carve((size_t)Etot * HEADS * 4);
    float*  acc   = (float*) carve((size_t)N * H1_F * 4);
    float*  logit = (float*) carve((size_t)N * OUT_F * 4);

    float* embOut  = (float*)d_out;                 // [N, 64]
    float* predOut = (float*)d_out + (size_t)N * OUT_F;

    const int B = 256;

    // one-time prep: edge list, weight packing, input conversion
    k_build_edges<<<gblk(Etot, B), B, 0, stream>>>(ei, srcF, dstF, E, N);
    k_pack_w_bf16<<<gblk((long long)IN_F * H1_F,  B), B, 0, stream>>>(W1, W1b, IN_F,  H1_F);
    k_pack_w_bf16<<<gblk((long long)H1_F * H1_F,  B), B, 0, stream>>>(W2, W2b, H1_F,  H1_F);
    k_pack_w_bf16<<<gblk((long long)H1_F * HID_F, B), B, 0, stream>>>(W3, W3b, H1_F,  HID_F);
    k_pack_w_bf16<<<gblk((long long)HID_F * OUT_F,B), B, 0, stream>>>(Wc, Wcb, HID_F, OUT_F);
    k_f32_to_bf16<<<gblk((long long)N * IN_F, B), B, 0, stream>>>(x, Abf, N * IN_F);

    auto gat_layer = [&](int Kin, const __bf16* Wb, const float* a_s, const float* a_d,
                         const float* bvec, int H, int C, float* outF32) {
        const int F = H * C;
        const long long tiles = (long long)((N + 15) / 16) * (F / 64);
        k_gemm_wmma_bf16<<<gblk(tiles, 4), 128, 0, stream>>>(Abf, Wb, nullptr, xw, N, Kin, F);
        k_scores<<<gblk((long long)N * H, B), B, 0, stream>>>(xw, a_s, a_d, sS, sD, N, H, C);
        k_fill_f32<<<gblk((long long)N * H, B), B, 0, stream>>>(mMax, -3.0e38f, N * H);
        k_fill_f32<<<gblk((long long)N * H, B), B, 0, stream>>>(den, 0.0f, N * H);
        k_fill_f32<<<gblk((long long)N * F, B), B, 0, stream>>>(acc, 0.0f, N * F);
        k_edge_alpha<<<gblk((long long)Etot * H, B), B, 0, stream>>>(srcF, dstF, sS, sD, alpha, mMax, Etot, H);
        k_edge_exp  <<<gblk((long long)Etot * H, B), B, 0, stream>>>(dstF, alpha, mMax, den, Etot, H);
        k_edge_scatter<<<gblk((long long)Etot * 32, B), B, 0, stream>>>(srcF, dstF, alpha, den, xw, acc, Etot, H, C);
        k_bias_relu<<<gblk((long long)N * F, B), B, 0, stream>>>(acc, bvec, outF32, Abf, N * F, F);
    };

    gat_layer(IN_F, W1b, as1, ad1, b1, HEADS, HID_F, nullptr);   // layer 1: 128 -> 4x64
    gat_layer(H1_F, W2b, as2, ad2, b2, HEADS, HID_F, nullptr);   // layer 2: 256 -> 4x64
    gat_layer(H1_F, W3b, as3, ad3, b3, 1,     HID_F, embOut);    // layer 3: 256 -> 64 (emb)

    // classifier: logits = emb @ Wc + bc ; preds = log_softmax(logits)
    {
        const long long tiles = (long long)((N + 15) / 16) * (OUT_F / 64);
        k_gemm_wmma_bf16<<<gblk(tiles, 4), 128, 0, stream>>>(Abf, Wcb, bc, logit, N, HID_F, OUT_F);
    }
    k_log_softmax<<<gblk((long long)N * 32, B), B, 0, stream>>>(logit, predOut, N, OUT_F);
}
